// MambaEncoder_15925738734299
// MI455X (gfx1250) — compile-verified, matrix operand order checked by reading, not run on hardware
//
#include <hip/hip_runtime.h>
#include <hip/hip_bf16.h>

// ---------------------------------------------------------------------------
// Problem constants (from reference)
// ---------------------------------------------------------------------------
#define BB     4
#define LL     256
#define C_IN   64
#define DD     1024
#define DEPTH  16
#define NN     16
#define RR     64
#define DCONV  4
#define HEADS  16
#define LQ     77
#define HD     64
#define OUT_F  768

#define BLD    (BB * LL * DD)         // 1,048,576
#define MTOK   (BB * LL)              // 1024 tokens
#define MHEAD  (BB * LQ)              // 308 rows in head

// ---------------------------------------------------------------------------
// WMMA GEMM:  C[M,N] = act( A[M,K] * W[N,K]^T + bias[N] )   (f32 in memory,
// bf16 operands / f32 accumulate via v_wmma_f32_16x16x32_bf16)
// K must be a multiple of 32 (true for every call site: 64/96->no! K in
// {64,1024,1536} only; N may be ragged).
// flags: bit0 = accumulate (+=), bit1 = reverse l within each L=256 segment
// act: 0 none, 1 silu, 2 softplus
// ---------------------------------------------------------------------------
typedef __attribute__((ext_vector_type(16))) __bf16 v16bf;
typedef __attribute__((ext_vector_type(8)))  float  v8f;

union FragU { v16bf v; uint4 q[2]; };

__device__ __forceinline__ unsigned short f32_to_bf16_bits(float f) {
  union { float f; unsigned u; } c; c.f = f;
  unsigned r = c.u + 0x7FFFu + ((c.u >> 16) & 1u);   // round-to-nearest-even
  return (unsigned short)(r >> 16);
}
__device__ __forceinline__ unsigned pack_bf16x2(float lo, float hi) {
  return ((unsigned)f32_to_bf16_bits(hi) << 16) | (unsigned)f32_to_bf16_bits(lo);
}
__device__ __forceinline__ float silu_f(float x) { return x / (1.0f + __expf(-x)); }

#define BM   128
#define BN   128
#define KT   32
#define LDT  40   // ushort row stride: 32 + 8 pad -> 80 bytes (16B aligned rows)
#define LDTU 20   // same stride in dwords

__global__ __launch_bounds__(256)
void k_gemm(const float* __restrict__ A, int lda,
            const float* __restrict__ W, int ldw,
            const float* __restrict__ bias,
            float* __restrict__ Cout, int ldc,
            int M, int Nn, int K, int act, int flags)
{
  __shared__ unsigned short As[BM * LDT];   // 10,240 B
  __shared__ unsigned short Bs[BN * LDT];   // 10,240 B

  const int tid     = threadIdx.x;
  const int lane    = tid & 31;
  const int wave    = tid >> 5;
  const int wm      = wave >> 1;            // 0..3  -> 32-row strip
  const int wn      = wave & 1;             // 0..1  -> 64-col strip
  const int block_m = blockIdx.y * BM;
  const int block_n = blockIdx.x * BN;

  v8f acc[2][4] = {};

  // loader mapping: 256 threads; tile is 128 rows x 32 cols; each thread
  // converts one 16-float half-row (4x float4 -> 2x uint4 of packed bf16)
  const int lrow  = tid >> 1;
  const int lhalf = tid & 1;
  const int lcol0 = lhalf * 16;

  // WMMA bf16 16x32 operand layout (ISA 7.12.2):
  // lanes 0-15: row=lane, K {0..7,16..23}; lanes 16-31: row=lane-16, K {8..15,24..31}
  const int fr = lane & 15;
  const int kh = (lane >> 4) * 8;

  unsigned* AsU = (unsigned*)As;
  unsigned* BsU = (unsigned*)Bs;

  for (int k0 = 0; k0 < K; k0 += KT) {
    // ---- stage A half-row ----
    {
      const int gm = block_m + lrow;
      uint4 lo = {0, 0, 0, 0}, hi = {0, 0, 0, 0};
      if (gm < M) {
        const float4* s4 = (const float4*)(A + (size_t)gm * lda + k0 + lcol0);
        if (k0 + KT < K) __builtin_prefetch((const float*)s4 + KT, 0, 1);
        float4 f0 = s4[0], f1 = s4[1], f2 = s4[2], f3 = s4[3];
        lo.x = pack_bf16x2(f0.x, f0.y); lo.y = pack_bf16x2(f0.z, f0.w);
        lo.z = pack_bf16x2(f1.x, f1.y); lo.w = pack_bf16x2(f1.z, f1.w);
        hi.x = pack_bf16x2(f2.x, f2.y); hi.y = pack_bf16x2(f2.z, f2.w);
        hi.z = pack_bf16x2(f3.x, f3.y); hi.w = pack_bf16x2(f3.z, f3.w);
      }
      uint4* dst = (uint4*)(AsU + lrow * LDTU + lhalf * 8);
      dst[0] = lo; dst[1] = hi;
    }
    // ---- stage W half-row ----
    {
      const int gn = block_n + lrow;
      uint4 lo = {0, 0, 0, 0}, hi = {0, 0, 0, 0};
      if (gn < Nn) {
        const float4* s4 = (const float4*)(W + (size_t)gn * ldw + k0 + lcol0);
        if (k0 + KT < K) __builtin_prefetch((const float*)s4 + KT, 0, 1);
        float4 f0 = s4[0], f1 = s4[1], f2 = s4[2], f3 = s4[3];
        lo.x = pack_bf16x2(f0.x, f0.y); lo.y = pack_bf16x2(f0.z, f0.w);
        lo.z = pack_bf16x2(f1.x, f1.y); lo.w = pack_bf16x2(f1.z, f1.w);
        hi.x = pack_bf16x2(f2.x, f2.y); hi.y = pack_bf16x2(f2.z, f2.w);
        hi.z = pack_bf16x2(f3.x, f3.y); hi.w = pack_bf16x2(f3.z, f3.w);
      }
      uint4* dst = (uint4*)(BsU + lrow * LDTU + lhalf * 8);
      dst[0] = lo; dst[1] = hi;
    }
    __syncthreads();

    // ---- fragments + 8 WMMA per wave per K-step ----
    FragU a[2];
    #pragma unroll
    for (int mt = 0; mt < 2; ++mt) {
      const unsigned short* base = &As[(wm * 32 + mt * 16 + fr) * LDT];
      a[mt].q[0] = *(const uint4*)(base + kh);
      a[mt].q[1] = *(const uint4*)(base + 16 + kh);
    }
    #pragma unroll
    for (int nt = 0; nt < 4; ++nt) {
      FragU b;
      const unsigned short* base = &Bs[(wn * 64 + nt * 16 + fr) * LDT];
      b.q[0] = *(const uint4*)(base + kh);
      b.q[1] = *(const uint4*)(base + 16 + kh);
      #pragma unroll
      for (int mt = 0; mt < 2; ++mt)
        acc[mt][nt] = __builtin_amdgcn_wmma_f32_16x16x32_bf16(
            false, a[mt].v, false, b.v, (short)0, acc[mt][nt], false, false);
    }
    __syncthreads();
  }

  // ---- epilogue: C/D layout col = lane&15, row = vgpr + 8*(lane>>4) ----
  const int col  = lane & 15;
  const int rsub = (lane >> 4) * 8;
  #pragma unroll
  for (int nt = 0; nt < 4; ++nt) {
    const int n = block_n + wn * 64 + nt * 16 + col;
    if (n >= Nn) continue;
    const float bval = bias ? bias[n] : 0.0f;
    #pragma unroll
    for (int mt = 0; mt < 2; ++mt) {
      const int mbase = block_m + wm * 32 + mt * 16 + rsub;
      #pragma unroll
      for (int v = 0; v < 8; ++v) {
        const int m = mbase + v;
        if (m >= M) continue;
        float x = acc[mt][nt][v] + bval;
        if (act == 1)      x = silu_f(x);
        else if (act == 2) x = (x > 20.0f) ? x : log1pf(__expf(x));   // softplus
        int mo = m;
        if (flags & 2) { int b = m >> 8; int l = m & 255; mo = (b << 8) + (255 - l); }
        float* p = Cout + (size_t)mo * ldc + n;
        if (flags & 1) *p += x; else *p = x;
      }
    }
  }
}

// ---------------------------------------------------------------------------
// Elementwise / helper kernels
// ---------------------------------------------------------------------------
__global__ void k_zero(float* __restrict__ p, int n) {
  int i = blockIdx.x * blockDim.x + threadIdx.x;
  if (i < n) p[i] = 0.0f;
}

__global__ void k_add(float* __restrict__ dst, const float* __restrict__ src, int n) {
  int i = blockIdx.x * blockDim.x + threadIdx.x;
  if (i < n) dst[i] += src[i];
}

__global__ void k_reverse(const float* __restrict__ in, float* __restrict__ out) {
  int i = blockIdx.x * blockDim.x + threadIdx.x;
  if (i >= BLD) return;
  int d = i & (DD - 1);
  int l = (i >> 10) & (LL - 1);
  int b = i >> 18;
  out[i] = in[((size_t)(b * LL + (LL - 1 - l))) * DD + d];
}

// LayerNorm over D=1024, one block per token
__global__ __launch_bounds__(256)
void k_ln(const float* __restrict__ x, const float* __restrict__ w,
          float* __restrict__ out)
{
  __shared__ float red[256];
  const int row = blockIdx.x;
  const float* xr = x + (size_t)row * DD;
  float lx[4]; float s = 0.0f;
  #pragma unroll
  for (int j = 0; j < 4; ++j) { lx[j] = xr[threadIdx.x + j * 256]; s += lx[j]; }
  red[threadIdx.x] = s; __syncthreads();
  for (int st = 128; st > 0; st >>= 1) {
    if (threadIdx.x < st) red[threadIdx.x] += red[threadIdx.x + st];
    __syncthreads();
  }
  const float mean = red[0] * (1.0f / DD);
  __syncthreads();
  float vs = 0.0f;
  #pragma unroll
  for (int j = 0; j < 4; ++j) { float d = lx[j] - mean; vs += d * d; }
  red[threadIdx.x] = vs; __syncthreads();
  for (int st = 128; st > 0; st >>= 1) {
    if (threadIdx.x < st) red[threadIdx.x] += red[threadIdx.x + st];
    __syncthreads();
  }
  const float rstd = rsqrtf(red[0] * (1.0f / DD) + 1e-5f);
  #pragma unroll
  for (int j = 0; j < 4; ++j) {
    int d = threadIdx.x + j * 256;
    out[(size_t)row * DD + d] = (lx[j] - mean) * rstd * w[d];
  }
}

// depthwise causal conv (DCONV=4) + bias + silu; x = first D cols of xz (stride 2D)
__global__ void k_conv(const float* __restrict__ xz, const float* __restrict__ w4,
                       const float* __restrict__ cb, float* __restrict__ out)
{
  int i = blockIdx.x * blockDim.x + threadIdx.x;
  if (i >= BLD) return;
  int d = i & (DD - 1);
  int l = (i >> 10) & (LL - 1);
  int b = i >> 18;
  float s = cb[d];
  #pragma unroll
  for (int j = 0; j < DCONV; ++j) {
    int lp = l - (DCONV - 1) + j;
    if (lp >= 0) s += w4[d * DCONV + j] * xz[((size_t)(b * LL + lp)) * (2 * DD) + d];
  }
  out[i] = silu_f(s);
}

// selective scan: thread = (b,d), N=16 state in registers; B_t/C_t via LDS
// epilogue fuses  (y + x*Dp) * silu(z)
__global__ __launch_bounds__(256)
void k_scan(const float* __restrict__ dt, const float* __restrict__ xc,
            const float* __restrict__ xz,   // z at column offset D, row stride 2D
            const float* __restrict__ xdbl, // B at +R, C at +R+N, row stride R+2N
            const float* __restrict__ Alog, const float* __restrict__ Dp,
            float* __restrict__ y)
{
  __shared__ float sB[NN], sC[NN];
  const int b = blockIdx.x >> 2;
  const int d = ((blockIdx.x & 3) << 8) + threadIdx.x;

  float Ad[NN];
  #pragma unroll
  for (int n = 0; n < NN; ++n) Ad[n] = -__expf(Alog[d * NN + n]);
  const float dpv = Dp[d];
  float h[NN];
  #pragma unroll
  for (int n = 0; n < NN; ++n) h[n] = 0.0f;

  for (int l = 0; l < LL; ++l) {
    const size_t row = (size_t)(b * LL + l);
    if (threadIdx.x < 2 * NN) {
      float vv = xdbl[row * (RR + 2 * NN) + RR + threadIdx.x];
      if (threadIdx.x < NN) sB[threadIdx.x] = vv; else sC[threadIdx.x - NN] = vv;
    }
    __syncthreads();
    const float dtv = dt[row * DD + d];
    const float xv  = xc[row * DD + d];
    const float zv  = xz[row * (2 * DD) + DD + d];
    const float dx  = dtv * xv;
    float yv = 0.0f;
    #pragma unroll
    for (int n = 0; n < NN; ++n) {
      h[n] = __expf(dtv * Ad[n]) * h[n] + dx * sB[n];
      yv += h[n] * sC[n];
    }
    y[row * DD + d] = (yv + xv * dpv) * silu_f(zv);
    __syncthreads();
  }
}

// cross attention: one block per (lq, head, b); K=256 keys, HD=64
__global__ __launch_bounds__(256)
void k_attn(const float* __restrict__ q, const float* __restrict__ kv,
            float* __restrict__ o)
{
  __shared__ float qs[HD];
  __shared__ float att[LL];
  __shared__ float red[256];
  const int lq = blockIdx.x, hh = blockIdx.y, b = blockIdx.z;
  const int t = threadIdx.x;
  if (t < HD) qs[t] = q[((size_t)hh * LQ + lq) * HD + t];
  __syncthreads();

  const float* krow = kv + ((size_t)(b * LL + t)) * (2 * DD) + hh * (2 * HD);
  float s = 0.0f;
  #pragma unroll 8
  for (int dd = 0; dd < HD; ++dd) s += qs[dd] * krow[dd];
  s *= 0.125f;                                  // 1/sqrt(64)
  red[t] = s; __syncthreads();
  for (int st = 128; st > 0; st >>= 1) {
    if (t < st) red[t] = fmaxf(red[t], red[t + st]);
    __syncthreads();
  }
  const float mx = red[0]; __syncthreads();
  const float e = __expf(s - mx);
  att[t] = e; red[t] = e; __syncthreads();
  for (int st = 128; st > 0; st >>= 1) {
    if (t < st) red[t] += red[t + st];
    __syncthreads();
  }
  const float inv = 1.0f / red[0];
  __syncthreads();
  if (t < HD) {
    float acc = 0.0f;
    for (int kk = 0; kk < LL; ++kk)
      acc += att[kk] * kv[((size_t)(b * LL + kk)) * (2 * DD) + hh * (2 * HD) + HD + t];
    o[((size_t)(b * LQ + lq)) * DD + hh * HD + t] = acc * inv;
  }
}

// split (B,LQ,1536) -> a=(...,:768) then b=(...,768:) concatenated flat
__global__ void k_split(const float* __restrict__ s, float* __restrict__ out) {
  int i = blockIdx.x * blockDim.x + threadIdx.x;
  if (i >= MHEAD * 2 * OUT_F) return;
  int m = i / (2 * OUT_F), n = i % (2 * OUT_F);
  float v = s[i];
  if (n < OUT_F) out[(size_t)m * OUT_F + n] = v;
  else out[(size_t)MHEAD * OUT_F + (size_t)m * OUT_F + (n - OUT_F)] = v;
}

// ---------------------------------------------------------------------------
// Host launch
// ---------------------------------------------------------------------------
static inline void gemm(hipStream_t st, const float* A, int lda, const float* W,
                        int ldw, const float* bias, float* C, int ldc,
                        int M, int Nn, int K, int act, int flags)
{
  dim3 g((Nn + BN - 1) / BN, (M + BM - 1) / BM);
  k_gemm<<<g, 256, 0, st>>>(A, lda, W, ldw, bias, C, ldc, M, Nn, K, act, flags);
}

extern "C" void kernel_launch(void* const* d_in, const int* in_sizes, int n_in,
                              void* d_out, int out_size, void* d_ws, size_t ws_size,
                              hipStream_t stream)
{
  const float* x      = (const float*)d_in[0];
  const float* stem_W = (const float*)d_in[1];
  const float* stem_b = (const float*)d_in[2];
  const float* norm_w = (const float*)d_in[3];
  const float* in_W   = (const float*)d_in[4];
  const float* conv_w = (const float*)d_in[5];
  const float* conv_b = (const float*)d_in[6];
  const float* xprojW = (const float*)d_in[7];
  const float* dt_W   = (const float*)d_in[8];
  const float* dt_b   = (const float*)d_in[9];
  const float* A_log  = (const float*)d_in[10];
  const float* D_p    = (const float*)d_in[11];
  const float* out_W  = (const float*)d_in[12];
  const float* qq     = (const float*)d_in[13];
  const float* kv_W   = (const float*)d_in[14];
  const float* po_W   = (const float*)d_in[15];
  const float* po_b   = (const float*)d_in[16];
  const float* h1_W   = (const float*)d_in[17];
  const float* h1_b   = (const float*)d_in[18];
  const float* h2_W   = (const float*)d_in[19];
  const float* h2_b   = (const float*)d_in[20];
  float* outp = (float*)d_out;

  // workspace carve
  size_t off = 0;
  auto carve = [&](size_t elems) {
    float* p = (float*)((char*)d_ws + off);
    off += (elems * sizeof(float) + 255) & ~(size_t)255;
    return p;
  };
  float* res   = carve(BLD);
  float* hbuf  = carve(BLD);
  float* hn    = carve(BLD);
  float* hnrev = carve(BLD);
  float* xzb   = carve((size_t)MTOK * 2 * DD);
  float* xconv = carve(BLD);
  float* xdbl  = carve((size_t)MTOK * (RR + 2 * NN));
  float* dtbuf = carve(BLD);
  float* ybuf  = carve(BLD);
  float* kvb   = carve((size_t)MTOK * 2 * DD);
  float* obuf  = carve((size_t)MHEAD * DD);
  float* pob   = carve((size_t)MHEAD * DD);
  float* s1    = carve((size_t)MHEAD * 2 * OUT_F);
  float* s2    = carve((size_t)MHEAD * 2 * OUT_F);
  (void)ws_size; (void)in_sizes; (void)n_in; (void)out_size;

  const int EW = (BLD + 255) / 256;

  // stem: h = x @ stem_W^T + stem_b
  gemm(stream, x, C_IN, stem_W, C_IN, stem_b, hbuf, DD, MTOK, DD, C_IN, 0, 0);
  k_zero<<<EW, 256, 0, stream>>>(res, BLD);

  for (int i = 0; i < DEPTH; ++i) {
    k_add<<<EW, 256, 0, stream>>>(res, hbuf, BLD);          // res += h
    k_ln<<<MTOK, 256, 0, stream>>>(res, norm_w + (size_t)i * DD, hn);
    k_reverse<<<EW, 256, 0, stream>>>(hn, hnrev);

    for (int dir = 0; dir < 2; ++dir) {
      const size_t s2d = (size_t)(i * 2 + dir);
      const float* U = dir ? hnrev : hn;
      // xz = U @ in_W^T       (1024 -> 2048)
      gemm(stream, U, DD, in_W + s2d * 2 * DD * DD, DD, nullptr,
           xzb, 2 * DD, MTOK, 2 * DD, DD, 0, 0);
      // depthwise causal conv + silu
      k_conv<<<EW, 256, 0, stream>>>(xzb, conv_w + s2d * DD * DCONV,
                                     conv_b + s2d * DD, xconv);
      // xdbl = xconv @ xproj_W^T   (1024 -> 96)
      gemm(stream, xconv, DD, xprojW + s2d * (RR + 2 * NN) * DD, DD, nullptr,
           xdbl, RR + 2 * NN, MTOK, RR + 2 * NN, DD, 0, 0);
      // dt = softplus(xdbl[:, :R] @ dt_W^T + dt_b)  (K=64), fused bias+softplus
      gemm(stream, xdbl, RR + 2 * NN, dt_W + s2d * DD * RR, RR,
           dt_b + s2d * DD, dtbuf, DD, MTOK, DD, RR, 2, 0);
      // selective scan (+ x*Dp, * silu(z) fused)
      k_scan<<<16, 256, 0, stream>>>(dtbuf, xconv, xzb, xdbl,
                                     A_log + s2d * DD * NN, D_p + s2d * DD, ybuf);
      // out-proj; dir==1 fuses reversed accumulate into hbuf
      gemm(stream, ybuf, DD, out_W + s2d * DD * DD, DD, nullptr,
           hbuf, DD, MTOK, DD, DD, 0, dir == 0 ? 0 : 3);
    }
  }

  // final residual + LN
  k_add<<<EW, 256, 0, stream>>>(res, hbuf, BLD);
  k_ln<<<MTOK, 256, 0, stream>>>(res, norm_w + (size_t)DEPTH * DD, hn);

  // kv projection (1024 -> 2048)
  gemm(stream, hn, DD, kv_W, DD, nullptr, kvb, 2 * DD, MTOK, 2 * DD, DD, 0, 0);

  // cross attention
  k_attn<<<dim3(LQ, HEADS, BB), 256, 0, stream>>>(qq, kvb, obuf);

  // head: po (+bias, silu fused), h1 (+bias, silu), h2 (+bias)
  gemm(stream, obuf, DD, po_W, DD, po_b, pob, DD, MHEAD, DD, DD, 1, 0);
  gemm(stream, pob, DD, h1_W, DD, h1_b, s1, 2 * OUT_F, MHEAD, 2 * OUT_F, DD, 1, 0);
  gemm(stream, s1, 2 * OUT_F, h2_W, 2 * OUT_F, h2_b, s2, 2 * OUT_F,
       MHEAD, 2 * OUT_F, 2 * OUT_F, 0, 0);

  // split into (a, b) concatenated flat
  k_split<<<(MHEAD * 2 * OUT_F + 255) / 256, 256, 0, stream>>>(s2, outp);
}
